// ChebyNet_38070590112102
// MI455X (gfx1250) — matrix operand order repaired and checked by reading, not compiled, not run on hardware
//
#include <hip/hip_runtime.h>

typedef __attribute__((ext_vector_type(16))) __bf16 v16bf;
typedef __attribute__((ext_vector_type(8)))  float  v8f;
typedef __attribute__((ext_vector_type(16))) float  v16f;

#define IN_CH 1024
#define HID   64
#define ZD    32

union BF16V { v16bf v; unsigned short u[16]; };
static_assert(sizeof(BF16V) == 32, "BF16V must be 32 bytes");
union F16U { v16f v; float4 f4[4]; };

// ---------------- degree / normalization ----------------

__global__ void deg_kernel(const int* __restrict__ edges, float* __restrict__ deg,
                           int n_edges, int n_nodes) {
  int t = blockIdx.x * blockDim.x + threadIdx.x;
  int etot = n_edges + n_nodes;
  if (t >= etot) return;
  int d = (t < n_edges) ? edges[n_edges + t] : (t - n_edges);
  atomicAdd(&deg[d], 1.0f);
}

__global__ void dinv_kernel(float* __restrict__ deg, int n) {
  int t = blockIdx.x * blockDim.x + threadIdx.x;
  if (t >= n) return;
  float dg = deg[t];
  deg[t] = (dg > 0.0f) ? rsqrtf(dg) : 0.0f;
}

// ---------------- weight transpose + bf16 convert ----------------
// W is [K, N] row-major f32; Wt is [N, K] row-major bf16 (a column of the
// WMMA B-matrix == 16 contiguous bf16 per lane).

__global__ void cvt_w_kernel(const float* __restrict__ W, unsigned short* __restrict__ Wt,
                             int K, int N) {
  int t = blockIdx.x * blockDim.x + threadIdx.x;
  if (t >= K * N) return;
  int n = t / K, k = t - n * K;
  ((__bf16*)Wt)[n * K + k] = (__bf16)W[k * N + n];   // native v_cvt, RNE
}

// ---------------- GEMM1: h = x @ W1, [N,1024]x[1024,64] ----------------
// One wave per 16-row M tile, 4 accumulators covering all 64 output cols so
// x streams from HBM exactly once. W1^T is staged in LDS in two 64 KB phases
// via CDNA5 async global->LDS copies (ASYNCcnt), and A-tiles are register
// double-buffered so VMEM latency overlaps the WMMAs.

__global__ void __launch_bounds__(256) gemm1_kernel(const float* __restrict__ x,
                                                    const unsigned short* __restrict__ W1t,
                                                    float* __restrict__ h, int n_nodes) {
  extern __shared__ unsigned short sW[];     // [HID][512] bf16 = 64 KB per phase
  const int tid   = threadIdx.x;
  const int lane  = tid & 31;
  const int wave  = tid >> 5;
  const int mtile = blockIdx.x * 8 + wave;
  const int mtiles = (n_nodes + 15) >> 4;
  const bool active = (mtile < mtiles);      // inactive waves still run barriers/staging

  const int m0    = active ? (mtile << 4) : 0;
  const int lrow  = lane & 15;
  const int lhalf = lane >> 4;               // 0 or 1
  int m = m0 + lrow;
  if (m >= n_nodes) m = n_nodes - 1;         // per-lane clamp (cndmask, no divergence)
  const int koff  = lhalf * 16;              // A layout: halves hold K {koff..+15}
  const int kbase = lhalf * 16;              // B layout: lanes 16-31 hold K=16..31
  const float* xrow = x + (size_t)m * IN_CH;

  const unsigned lds_base = (unsigned)__builtin_amdgcn_groupstaticsize();

  v8f acc0 = {}, acc1 = {}, acc2 = {}, acc3 = {};

  // prime A double-buffer (K = 0..31 slice for this lane)
  float4 ca0 = *(const float4*)(xrow + koff);
  float4 ca1 = *(const float4*)(xrow + koff + 4);
  float4 ca2 = *(const float4*)(xrow + koff + 8);
  float4 ca3 = *(const float4*)(xrow + koff + 12);

  for (int kt = 0; kt < IN_CH; kt += 32) {
    if ((kt & 511) == 0) {                    // stage next 64 KB panel of W1^T
      if (kt != 0) __syncthreads();           // everyone done reading previous panel
      for (int i = tid; i < (HID * 512) / 8; i += 256) {
        int el = i * 8;
        int n  = el >> 9;                     // row (output col) 0..63
        int kk = el & 511;                    // K within panel
        const unsigned short* g = W1t + n * IN_CH + kt + kk;
        unsigned dst = lds_base + (unsigned)(i * 16);
        asm volatile("global_load_async_to_lds_b128 %0, %1, off"
                     :: "v"(dst), "v"(g) : "memory");
      }
      asm volatile("s_wait_asynccnt 0x0" ::: "memory");
      __syncthreads();
    }

    // prefetch next A slice while this iteration computes (clamped on last iter)
    const int ktn = (kt + 32 < IN_CH) ? (kt + 32) : kt;
    const float* pn = xrow + ktn + koff;
    float4 na0 = *(const float4*)(pn);
    float4 na1 = *(const float4*)(pn + 4);
    float4 na2 = *(const float4*)(pn + 8);
    float4 na3 = *(const float4*)(pn + 12);

    // convert current A slice with native packed cvt
    F16U t;
    t.f4[0] = ca0; t.f4[1] = ca1; t.f4[2] = ca2; t.f4[3] = ca3;
    v16bf af = __builtin_convertvector(t.v, v16bf);

    // B fragments from LDS (ds_load_b128 pairs)
    const unsigned short* pb = sW + (kt & 511) + kbase;
    v16bf b0 = ((const BF16V*)(pb + (0 * 16 + lrow) * 512))->v;
    v16bf b1 = ((const BF16V*)(pb + (1 * 16 + lrow) * 512))->v;
    v16bf b2 = ((const BF16V*)(pb + (2 * 16 + lrow) * 512))->v;
    v16bf b3 = ((const BF16V*)(pb + (3 * 16 + lrow) * 512))->v;

    acc0 = __builtin_amdgcn_wmma_f32_16x16x32_bf16(false, af, false, b0, (short)0, acc0, false, false);
    acc1 = __builtin_amdgcn_wmma_f32_16x16x32_bf16(false, af, false, b1, (short)0, acc1, false, false);
    acc2 = __builtin_amdgcn_wmma_f32_16x16x32_bf16(false, af, false, b2, (short)0, acc2, false, false);
    acc3 = __builtin_amdgcn_wmma_f32_16x16x32_bf16(false, af, false, b3, (short)0, acc3, false, false);

    ca0 = na0; ca1 = na1; ca2 = na2; ca3 = na3;
  }

  // C/D layout: VGPR r -> row m0 + lhalf*8 + r, col = lrow (+16*nt)
  if (active) {
    float* hp = h + (size_t)(m0 + lhalf * 8) * HID + lrow;
    if (m0 + 16 <= n_nodes) {                 // full-tile fast path, no per-row EXEC churn
      #pragma unroll
      for (int r = 0; r < 8; ++r) {
        hp[r * HID + 0]  = acc0[r];
        hp[r * HID + 16] = acc1[r];
        hp[r * HID + 32] = acc2[r];
        hp[r * HID + 48] = acc3[r];
      }
    } else {
      #pragma unroll
      for (int r = 0; r < 8; ++r) {
        if (m0 + lhalf * 8 + r < n_nodes) {
          hp[r * HID + 0]  = acc0[r];
          hp[r * HID + 16] = acc1[r];
          hp[r * HID + 32] = acc2[r];
          hp[r * HID + 48] = acc3[r];
        }
      }
    }
  }
}

// ---------------- edge scatter: agg[dst] += h[src] * dinv[src]*dinv[dst] ----------------
// h / agg are L2-resident (25.6 MB / 12.8 MB << 192 MB L2): float4 gathers +
// global_atomic_add_f32.

template <int CH>
__global__ void scatter_kernel(const float* __restrict__ h, const int* __restrict__ edges,
                               const float* __restrict__ dinv, float* __restrict__ agg,
                               int n_edges, int n_nodes) {
  const int G = CH / 4;
  int t = blockIdx.x * blockDim.x + threadIdx.x;
  int etot = n_edges + n_nodes;
  int e = t / G;
  if (e >= etot) return;
  int c = (t - e * G) * 4;
  int s, d;
  if (e < n_edges) { s = edges[e]; d = edges[n_edges + e]; }
  else             { s = d = e - n_edges; }
  float nrm = dinv[s] * dinv[d];
  float4 hv = *(const float4*)(h + (size_t)s * CH + c);
  float* ap = agg + (size_t)d * CH + c;
  atomicAdd(ap + 0, hv.x * nrm);
  atomicAdd(ap + 1, hv.y * nrm);
  atomicAdd(ap + 2, hv.z * nrm);
  atomicAdd(ap + 3, hv.w * nrm);
}

// ---------------- relu(agg1 + b1) -> bf16 ----------------

__global__ void relu_kernel(const float* __restrict__ agg, const float* __restrict__ b1,
                            unsigned short* __restrict__ hbf, int n) {
  int t = blockIdx.x * blockDim.x + threadIdx.x;
  if (t >= n) return;
  float v = agg[t] + b1[t & (HID - 1)];
  ((__bf16*)hbf)[t] = (__bf16)(v > 0.0f ? v : 0.0f);
}

// ---------------- GEMM2: h2 = h_bf16 @ W2, [N,64]x[64,32] ----------------

__global__ void __launch_bounds__(256) gemm2_kernel(const unsigned short* __restrict__ hbf,
                                                    const unsigned short* __restrict__ W2t,
                                                    float* __restrict__ h2, int n_nodes) {
  const int lane  = threadIdx.x & 31;
  const int wave  = threadIdx.x >> 5;
  const int mtile = blockIdx.x * 8 + wave;
  const int mtiles = (n_nodes + 15) >> 4;
  if (mtile >= mtiles) return;               // no barriers here: early return is safe

  const int m0    = mtile << 4;
  const int lrow  = lane & 15;
  const int lhalf = lane >> 4;
  int m = m0 + lrow;
  if (m >= n_nodes) m = n_nodes - 1;
  const int koff  = lhalf * 16;
  const int kbase = lhalf * 16;
  const unsigned short* arow = hbf + (size_t)m * HID;

  v8f acc0 = {}, acc1 = {};

  #pragma unroll
  for (int kt = 0; kt < HID; kt += 32) {
    BF16V af;
    ((uint4*)af.u)[0] = *(const uint4*)(arow + kt + koff);       // K = kt+koff..+7
    ((uint4*)af.u)[1] = *(const uint4*)(arow + kt + koff + 8);   // K = kt+koff+8..+15

    const unsigned short* pb = W2t + kt + kbase;                 // W2^T is 4 KB: L2/L0-resident
    v16bf b0 = ((const BF16V*)(pb + (0 * 16 + lrow) * HID))->v;
    v16bf b1 = ((const BF16V*)(pb + (1 * 16 + lrow) * HID))->v;

    acc0 = __builtin_amdgcn_wmma_f32_16x16x32_bf16(false, af.v, false, b0, (short)0, acc0, false, false);
    acc1 = __builtin_amdgcn_wmma_f32_16x16x32_bf16(false, af.v, false, b1, (short)0, acc1, false, false);
  }

  float* hp = h2 + (size_t)(m0 + lhalf * 8) * ZD + lrow;
  if (m0 + 16 <= n_nodes) {
    #pragma unroll
    for (int r = 0; r < 8; ++r) {
      hp[r * ZD + 0]  = acc0[r];
      hp[r * ZD + 16] = acc1[r];
    }
  } else {
    #pragma unroll
    for (int r = 0; r < 8; ++r) {
      if (m0 + lhalf * 8 + r < n_nodes) {
        hp[r * ZD + 0]  = acc0[r];
        hp[r * ZD + 16] = acc1[r];
      }
    }
  }
}

// ---------------- out init: broadcast b2 ----------------

__global__ void initout_kernel(float* __restrict__ out, const float* __restrict__ b2, int n) {
  int t = blockIdx.x * blockDim.x + threadIdx.x;
  if (t >= n) return;
  out[t] = b2[t & (ZD - 1)];
}

// ---------------- host launch ----------------

extern "C" void kernel_launch(void* const* d_in, const int* in_sizes, int n_in,
                              void* d_out, int out_size, void* d_ws, size_t ws_size,
                              hipStream_t stream) {
  const float* x     = (const float*)d_in[0];
  const int*   edges = (const int*)d_in[1];
  const float* W1    = (const float*)d_in[2];
  const float* b1    = (const float*)d_in[3];
  const float* W2    = (const float*)d_in[4];
  const float* b2    = (const float*)d_in[5];
  float* out = (float*)d_out;

  const int n_nodes = in_sizes[0] / IN_CH;
  const int n_edges = in_sizes[1] / 2;
  const int etot    = n_edges + n_nodes;

  char* ws = (char*)d_ws;
  size_t off = 0;
  auto alloc = [&](size_t bytes) -> void* {
    void* p = ws + off;
    off = (off + bytes + 255) & ~(size_t)255;
    return p;
  };
  float*          deg  = (float*)alloc((size_t)n_nodes * 4);
  float*          hlin = (float*)alloc((size_t)n_nodes * HID * 4);
  float*          agg1 = (float*)alloc((size_t)n_nodes * HID * 4);
  unsigned short* hbf  = (unsigned short*)alloc((size_t)n_nodes * HID * 2);
  unsigned short* W1t  = (unsigned short*)alloc((size_t)IN_CH * HID * 2);
  unsigned short* W2t  = (unsigned short*)alloc((size_t)HID * ZD * 2);
  float*          h2   = agg1;  // agg1 is dead after relu_kernel; reuse for layer-2 GEMM out

  hipMemsetAsync(deg,  0, (size_t)n_nodes * 4, stream);
  hipMemsetAsync(agg1, 0, (size_t)n_nodes * HID * 4, stream);

  deg_kernel<<<(etot + 255) / 256, 256, 0, stream>>>(edges, deg, n_edges, n_nodes);
  dinv_kernel<<<(n_nodes + 255) / 256, 256, 0, stream>>>(deg, n_nodes);

  cvt_w_kernel<<<(IN_CH * HID + 255) / 256, 256, 0, stream>>>(W1, W1t, IN_CH, HID);
  cvt_w_kernel<<<(HID * ZD + 255) / 256, 256, 0, stream>>>(W2, W2t, HID, ZD);

  const int mtiles = (n_nodes + 15) / 16;
  gemm1_kernel<<<(mtiles + 7) / 8, 256, (size_t)HID * 512 * 2, stream>>>(x, W1t, hlin, n_nodes);

  {
    long long th = (long long)etot * (HID / 4);
    scatter_kernel<HID><<<(unsigned)((th + 255) / 256), 256, 0, stream>>>(
        hlin, edges, deg, agg1, n_edges, n_nodes);
  }

  relu_kernel<<<((n_nodes * HID) + 255) / 256, 256, 0, stream>>>(agg1, b1, hbf, n_nodes * HID);

  gemm2_kernel<<<(mtiles + 7) / 8, 256, 0, stream>>>(hbf, W2t, h2, n_nodes);

  initout_kernel<<<((n_nodes * ZD) + 255) / 256, 256, 0, stream>>>(out, b2, n_nodes * ZD);

  {
    long long th = (long long)etot * (ZD / 4);
    scatter_kernel<ZD><<<(unsigned)((th + 255) / 256), 256, 0, stream>>>(
        h2, edges, deg, out, n_edges, n_nodes);
  }
}
